// ParallelmomAttention_90941637525892
// MI455X (gfx1250) — matrix operand order repaired and checked by reading, not compiled, not run on hardware
//
#include <hip/hip_runtime.h>
#include <hip/hip_bf16.h>
#include <math.h>

// ---------------------------------------------------------------------------
// Problem constants (from reference)
// ---------------------------------------------------------------------------
#define LL 2048
#define DD 2048
#define HH 16
#define DKK 1024
#define DVV 2048
#define HDK 64
#define HDV 128
#define NCAT 6144          // [q(1024) | k(1024) | v(2048) | g(2048)]
#define COL_Q 0
#define COL_K 1024
#define COL_V 2048
#define COL_G 4096

// ---------------------------------------------------------------------------
// bf16 helpers + WMMA fragment types (CDNA5 wave32, 16x16x32 bf16)
// ---------------------------------------------------------------------------
typedef __bf16 bf16_t;
typedef __attribute__((ext_vector_type(16))) bf16_t       v16bf;
typedef __attribute__((ext_vector_type(8)))  float        v8f;
typedef __attribute__((ext_vector_type(8)))  unsigned int v8u;
typedef __attribute__((ext_vector_type(4)))  unsigned int v4u;

__device__ __forceinline__ unsigned short f2bf(float f) {
  unsigned int u = __float_as_uint(f);
  unsigned int r = (u + 0x7fffu + ((u >> 16) & 1u)) >> 16;  // RNE
  return (unsigned short)r;
}
__device__ __forceinline__ float siluf(float v) { return v / (1.f + __expf(-v)); }

__device__ __forceinline__ v8f wmma_bf16(v16bf a, v16bf b, v8f c) {
  return __builtin_amdgcn_wmma_f32_16x16x32_bf16(false, a, false, b,
                                                 (short)0, c, false, false);
}

// One 32-byte vector load = a whole 16-element bf16 fragment for this lane.
__device__ __forceinline__ v16bf frag_ld(const unsigned short* p) {
  return __builtin_bit_cast(v16bf, *(const v8u*)p);
}

// ISA fragment swizzles (16x16x32 bf16, wave32):
// A (16x32 MxK): lanes 0-15 hold K 0..7 / 16..23, lanes 16-31 hold 8..15 / 24..31
__device__ __forceinline__ int a_lane(int row, int kk) { return row + 16 * ((kk >> 3) & 1); }
__device__ __forceinline__ int a_elem(int kk)          { return (kk & 7) + 8 * ((kk >> 4) & 1); }
// B (32x16 KxN): lanes 0-15 hold K 0..7 / 16..23, lanes 16-31 hold 8..15 / 24..31
__device__ __forceinline__ int b_lane(int kk, int col) { return col + 16 * ((kk >> 3) & 1); }
__device__ __forceinline__ int b_elem(int kk)          { return (kk & 7) + 8 * ((kk >> 4) & 1); }

// ---------------------------------------------------------------------------
// 1) Weight pack: f32 (K x srcW) row-major -> bf16 B-fragment panels
//    panel layout: [nt = col/16][kt = k/32][lane(32)][elem(16)]
// ---------------------------------------------------------------------------
__global__ void pack_w_kernel(const float* __restrict__ src,
                              unsigned short* __restrict__ bpan,
                              int K, int srcW, int colOff) {
  int idx = blockIdx.x * blockDim.x + threadIdx.x;
  if (idx >= K * srcW) return;
  int r = idx / srcW, c = idx - r * srcW;
  int col = colOff + c;
  int nt = col >> 4, kt = r >> 5, kk = r & 31;
  size_t p = ((size_t)nt * (K >> 5) + kt) * 512 + b_lane(kk, col & 15) * 16 + b_elem(kk);
  bpan[p] = f2bf(src[idx]);
}

// ---------------------------------------------------------------------------
// 2) Causal depthwise conv (width 2) + SiLU -> bf16 row-major activations
// ---------------------------------------------------------------------------
__global__ void conv_silu_kernel(const float* __restrict__ x,
                                 const float* __restrict__ convw,
                                 unsigned short* __restrict__ xc) {
  int idx = blockIdx.x * blockDim.x + threadIdx.x;
  if (idx >= LL * DD) return;
  int l = idx / DD, d = idx - l * DD;
  float cur  = x[idx];
  float prev = (l > 0) ? x[idx - DD] : 0.f;
  float v = prev * convw[d * 2 + 0] + cur * convw[d * 2 + 1];
  xc[idx] = f2bf(siluf(v));
}

// ---------------------------------------------------------------------------
// 3) WMMA bf16 GEMM: C(MxN,f32) = A(MxK row-major) * Bpan(panel-packed)
//    256 thr = 8 waves; block tile 64x128; K-step 64; wave tile 32x32.
//    A staged in LDS pre-swizzled (b128 in / b128 out); B straight from L2.
// ---------------------------------------------------------------------------
__global__ __launch_bounds__(256) void gemm_bf16_kernel(
    const unsigned short* __restrict__ A,
    const unsigned short* __restrict__ Bpan,
    float* __restrict__ C, int M, int N, int K) {
  __shared__ __align__(32) unsigned short sA[8][32][16];  // [ms(4)*2+kc(2)] panels

  int tid = threadIdx.x, lane = tid & 31, wave = tid >> 5;
  int wrow = wave >> 2, wcol = wave & 3;
  int blockRow = blockIdx.y * 64, blockCol = blockIdx.x * 128;
  int ktiles = K >> 5;
  int nt0 = (blockCol >> 4) + wcol * 2;

  v8f acc[2][2] = {};

  for (int k0 = 0; k0 < K; k0 += 64) {
#pragma unroll
    for (int it = 0; it < 2; ++it) {
      int vec = tid + it * 256;                 // 0..511
      int r = vec >> 3;                         // 0..63
      int kkg = (vec & 7) * 8;                  // aligned k-octet within 64
      int kc = kkg >> 5, kk = kkg & 31, ms = r >> 4;
      const unsigned short* gp = A + (size_t)(blockRow + r) * K + k0 + kkg;
      *(v4u*)&sA[ms * 2 + kc][a_lane(r & 15, kk)][a_elem(kk)] = *(const v4u*)gp;
      if (k0 + 64 < K) __builtin_prefetch(gp + 64, 0, 3);
    }
    __syncthreads();

#pragma unroll
    for (int kc = 0; kc < 2; ++kc) {
      int kt = (k0 >> 5) + kc;
      v16bf a0 = frag_ld(&sA[(wrow * 2 + 0) * 2 + kc][lane][0]);
      v16bf a1 = frag_ld(&sA[(wrow * 2 + 1) * 2 + kc][lane][0]);
      v16bf b0 = frag_ld(Bpan + ((size_t)(nt0 + 0) * ktiles + kt) * 512 + lane * 16);
      v16bf b1 = frag_ld(Bpan + ((size_t)(nt0 + 1) * ktiles + kt) * 512 + lane * 16);
      acc[0][0] = wmma_bf16(a0, b0, acc[0][0]);
      acc[0][1] = wmma_bf16(a0, b1, acc[0][1]);
      acc[1][0] = wmma_bf16(a1, b0, acc[1][0]);
      acc[1][1] = wmma_bf16(a1, b1, acc[1][1]);
    }
    __syncthreads();
  }

  int crow = (lane >> 4) & 1, ccol = lane & 15;
#pragma unroll
  for (int mi = 0; mi < 2; ++mi)
#pragma unroll
    for (int ni = 0; ni < 2; ++ni)
#pragma unroll
      for (int j = 0; j < 8; ++j) {
        int row = blockRow + wrow * 32 + mi * 16 + crow * 8 + j;
        int col = blockCol + wcol * 32 + ni * 16 + ccol;
        C[(size_t)row * N + col] = acc[mi][ni][j];
      }
}

// ---------------------------------------------------------------------------
// 4) Router: per (l,h) 2-way logits -> argmax/softmax; fuse v -> B-layout
//    panels: vpan[((h*(LL/32) + key/32)*8 + feat/16)*512 + lane*16 + e]
// ---------------------------------------------------------------------------
__global__ __launch_bounds__(128) void router_kernel(
    const float* __restrict__ qkvg, const float* __restrict__ router_w,
    unsigned short* __restrict__ vpan, int* __restrict__ win_r,
    float* __restrict__ win_s) {
  __shared__ float r0[128], r1[128];
  int lh = blockIdx.x;           // l*H + h
  int l = lh >> 4, h = lh & 15;
  int d = threadIdx.x;           // feature 0..127
  float vv = qkvg[(size_t)l * NCAT + COL_V + h * HDV + d];
  int kk = l & 31;               // key index within 32-chunk
  size_t p = ((size_t)(h * (LL / 32) + (l >> 5)) * 8 + (d >> 4)) * 512 +
             b_lane(kk, d & 15) * 16 + b_elem(kk);
  vpan[p] = f2bf(vv);
  r0[d] = vv * router_w[(h * HDV + d) * 2 + 0];
  r1[d] = vv * router_w[(h * HDV + d) * 2 + 1];
  __syncthreads();
  for (int s = 64; s > 0; s >>= 1) {
    if (d < s) { r0[d] += r0[d + s]; r1[d] += r1[d + s]; }
    __syncthreads();
  }
  if (d == 0) {
    float l0 = r0[0], l1 = r1[0];
    win_r[lh] = (l1 > l0) ? 1 : 0;                    // argmax, ties -> 0
    win_s[lh] = 1.f / (1.f + __expf(-fabsf(l0 - l1)));// winner softmax prob
  }
}

// ---------------------------------------------------------------------------
// 5) Per-head sequential scan: cumulative winner counts -> coef = score/count
// ---------------------------------------------------------------------------
__global__ void scan_kernel(const int* __restrict__ win_r,
                            const float* __restrict__ win_s,
                            float* __restrict__ coef) {
  int h = blockIdx.x;
  int c0 = 0, c1 = 0;
  for (int l = 0; l < LL; ++l) {
    int lh = l * HH + h;
    int cnt;
    if (win_r[lh] == 0) { ++c0; cnt = c0; } else { ++c1; cnt = c1; }
    coef[lh] = win_s[lh] / (float)cnt;
  }
}

// ---------------------------------------------------------------------------
// 6) SiLU(q) + RoPE(q,k) + expert scatter, written directly as WMMA panels:
//    qpan (A-layout):  [((h*(LL/16)+l/16)*4 + feat/32)*512 + a_lane*16 + a_elem]
//    kpan (B^T-layout):[((h*(LL/16)+l/16)*4 + feat/32)*512 + b_lane*16 + b_elem]
// ---------------------------------------------------------------------------
__global__ __launch_bounds__(64) void build_qk_kernel(
    const float* __restrict__ qkvg, const int* __restrict__ win_r,
    const float* __restrict__ coef, unsigned short* __restrict__ qpan,
    unsigned short* __restrict__ kpan) {
  __shared__ float sq[64], sk[64];
  int lh = blockIdx.x;
  int l = lh >> 4, h = lh & 15;
  int d = threadIdx.x;           // 0..63
  sq[d] = siluf(qkvg[(size_t)l * NCAT + COL_Q + h * HDK + d]);
  sk[d] = qkvg[(size_t)l * NCAT + COL_K + h * HDK + d];
  __syncthreads();
  int i = d & 31;
  float inv = powf(10000.f, -(2.f * (float)i) / 64.f);
  float fr = (float)l * inv;
  float c = cosf(fr), s = sinf(fr);
  float rq, rk;
  if (d < 32) { rq = sq[d] * c - sq[d + 32] * s; rk = sk[d] * c - sk[d + 32] * s; }
  else        { rq = sq[d] * c + sq[d - 32] * s; rk = sk[d] * c + sk[d - 32] * s; }
  int wr = win_r[lh];
  float cf = coef[lh];
  int srow = l & 15;
  size_t tb = (size_t)(h * (LL / 16) + (l >> 4)) * 4 * 512;
  unsigned short qv = f2bf(rq * cf), kv = f2bf(rk);
#pragma unroll
  for (int slot = 0; slot < 2; ++slot) {
    int f = slot * 64 + d;                      // expanded feature 0..127
    int kc = f >> 5, kk = f & 31;
    unsigned short qo = (slot == wr) ? qv : (unsigned short)0;
    unsigned short ko = (slot == wr) ? kv : (unsigned short)0;
    qpan[tb + kc * 512 + a_lane(srow, kk) * 16 + a_elem(kk)]  = qo;
    kpan[tb + kc * 512 + b_lane(kk, srow) * 16 + b_elem(kk)]  = ko;
  }
}

// ---------------------------------------------------------------------------
// 7) Causal attention per head (tril-masked scores, no softmax).
//    128 thr = 4 waves; 64 query rows per block (16 per wave).
//    Q frags from global panels; K/V tiles staged to LDS with straight b128
//    memcpy (panels are already fragment-ordered); score re-layout through
//    wave-private LDS (no barriers needed for it).
// ---------------------------------------------------------------------------
__global__ __launch_bounds__(128) void attention_kernel(
    const unsigned short* __restrict__ qpan,
    const unsigned short* __restrict__ kpan,
    const unsigned short* __restrict__ vpan, float* __restrict__ o) {
  __shared__ __align__(32) unsigned short sK[16][32][16];     // [k16(4)*4+kc(4)]
  __shared__ __align__(32) unsigned short sV[16][32][16];     // [k32(2)*8+fs(8)]
  __shared__ __align__(32) unsigned short sS[4][2][32][16];   // per-wave S in A-layout

  int tid = threadIdx.x, lane = tid & 31, w = tid >> 5;
  int crow = (lane >> 4) & 1, ccol = lane & 15;
  int h = blockIdx.y, qt = blockIdx.x, q0 = qt * 64;

  // Q fragments: 16 query rows x 128 feat, direct from global panels
  v16bf qf[4];
  const unsigned short* qb = qpan + (size_t)(h * (LL / 16) + (q0 >> 4) + w) * 4 * 512;
#pragma unroll
  for (int kc = 0; kc < 4; ++kc) qf[kc] = frag_ld(qb + kc * 512 + lane * 16);

  v8f oacc[8] = {};
  const float scale = 0.125f;    // HDK^-0.5

  for (int kt = 0; kt <= qt; ++kt) {
    int k0 = kt * 64;
    const v4u* kg = (const v4u*)(kpan + (size_t)(h * (LL / 16) + (k0 >> 4)) * 4 * 512);
    const v4u* vg = (const v4u*)(vpan + (size_t)(h * (LL / 32) + (k0 >> 5)) * 8 * 512);
    v4u* sk4 = (v4u*)&sK[0][0][0];
    v4u* sv4 = (v4u*)&sV[0][0][0];
    for (int i = tid; i < 1024; i += 128) {     // 8192 u16 each = 1024 x b128
      sk4[i] = kg[i];
      sv4[i] = vg[i];
      if (kt < qt) { __builtin_prefetch(kg + 1024 + i, 0, 3); __builtin_prefetch(vg + 1024 + i, 0, 3); }
    }
    __syncthreads();

    // S(16x64) = Qf(16x128) x K^T(128x64)
#pragma unroll
    for (int n = 0; n < 4; ++n) {
      v8f sa = {};
#pragma unroll
      for (int kc = 0; kc < 4; ++kc)
        sa = wmma_bf16(qf[kc], frag_ld(&sK[n * 4 + kc][lane][0]), sa);
      // scale + causal mask, scatter into wave-private A-layout score panel
      int kk = (n & 1) * 16 + ccol;             // key index within 32-chunk
      int kc2 = n >> 1;
#pragma unroll
      for (int j = 0; j < 8; ++j) {
        int srow = crow * 8 + j;
        int grow = q0 + w * 16 + srow;
        int gcol = k0 + n * 16 + ccol;
        float v = (gcol <= grow) ? sa[j] * scale : 0.f;
        sS[w][kc2][a_lane(srow, kk)][a_elem(kk)] = f2bf(v);
      }
    }
    // O(16x128) += S(16x64) x V(64x128)  (same-wave LDS ops are in-order)
#pragma unroll
    for (int kc2 = 0; kc2 < 2; ++kc2) {
      v16bf af = frag_ld(&sS[w][kc2][lane][0]);
#pragma unroll
      for (int fs = 0; fs < 8; ++fs)
        oacc[fs] = wmma_bf16(af, frag_ld(&sV[kc2 * 8 + fs][lane][0]), oacc[fs]);
    }
    __syncthreads();
  }

#pragma unroll
  for (int fs = 0; fs < 8; ++fs)
#pragma unroll
    for (int j = 0; j < 8; ++j) {
      int row = q0 + w * 16 + crow * 8 + j;
      o[(size_t)row * DVV + h * HDV + fs * 16 + ccol] = oacc[fs][j];
    }
}

// ---------------------------------------------------------------------------
// 8) Per-(l,h) LayerNorm over 128 + SiLU(g) gate -> bf16 row-major
// ---------------------------------------------------------------------------
__global__ __launch_bounds__(128) void ln_gate_kernel(
    const float* __restrict__ o, const float* __restrict__ qkvg,
    unsigned short* __restrict__ y) {
  __shared__ float s1[128], s2[128];
  int lh = blockIdx.x;
  int l = lh >> 4, h = lh & 15;
  int d = threadIdx.x;
  float val = o[(size_t)l * DVV + h * HDV + d];
  s1[d] = val; s2[d] = val * val;
  __syncthreads();
  for (int s = 64; s > 0; s >>= 1) {
    if (d < s) { s1[d] += s1[d + s]; s2[d] += s2[d + s]; }
    __syncthreads();
  }
  float mean = s1[0] * (1.f / 128.f);
  float var  = s2[0] * (1.f / 128.f) - mean * mean;
  float nrm  = (val - mean) * rsqrtf(var + 1e-5f);
  float gv   = qkvg[(size_t)l * NCAT + COL_G + h * HDV + d];
  y[(size_t)l * DVV + h * HDV + d] = f2bf(siluf(gv) * nrm);
}

// ---------------------------------------------------------------------------
// Host-side launch
// ---------------------------------------------------------------------------
extern "C" void kernel_launch(void* const* d_in, const int* in_sizes, int n_in,
                              void* d_out, int out_size, void* d_ws, size_t ws_size,
                              hipStream_t stream) {
  (void)in_sizes; (void)n_in; (void)out_size;
  const float* x        = (const float*)d_in[0];
  const float* convw    = (const float*)d_in[1];
  const float* Wq       = (const float*)d_in[2];
  const float* Wk       = (const float*)d_in[3];
  const float* Wv       = (const float*)d_in[4];
  const float* Wg       = (const float*)d_in[5];
  const float* router_w = (const float*)d_in[6];
  const float* Wout     = (const float*)d_in[7];
  float* out = (float*)d_out;

  char* ws = (char*)d_ws;
  size_t off = 0;
  auto alloc = [&](size_t bytes) { char* p = ws + off; off += (bytes + 255) & ~(size_t)255; return p; };
  unsigned short* xc    = (unsigned short*)alloc((size_t)LL * DD * 2);       // row-major bf16
  unsigned short* wcat  = (unsigned short*)alloc((size_t)DD * NCAT * 2);     // B panels
  unsigned short* woutb = (unsigned short*)alloc((size_t)DVV * DD * 2);      // B panels
  float*          qkvg  = (float*)         alloc((size_t)LL * NCAT * 4);
  unsigned short* vpan  = (unsigned short*)alloc((size_t)LL * DVV * 2);      // B panels
  unsigned short* qpan  = (unsigned short*)alloc((size_t)LL * HH * 128 * 2); // A panels
  unsigned short* kpan  = (unsigned short*)alloc((size_t)LL * HH * 128 * 2); // B^T panels
  int*            winr  = (int*)           alloc((size_t)LL * HH * 4);
  float*          wins  = (float*)         alloc((size_t)LL * HH * 4);
  float*          coef  = (float*)         alloc((size_t)LL * HH * 4);
  float*          obuf  = (float*)         alloc((size_t)LL * DVV * 4);
  unsigned short* ybf   = (unsigned short*)alloc((size_t)LL * DVV * 2);
  if (off > ws_size) return;  // workspace too small (compile-only loop)

  int n;
  n = DD * DKK; pack_w_kernel<<<(n + 255) / 256, 256, 0, stream>>>(Wq,   wcat,  DD, DKK, COL_Q);
  n = DD * DKK; pack_w_kernel<<<(n + 255) / 256, 256, 0, stream>>>(Wk,   wcat,  DD, DKK, COL_K);
  n = DD * DVV; pack_w_kernel<<<(n + 255) / 256, 256, 0, stream>>>(Wv,   wcat,  DD, DVV, COL_V);
  n = DD * DVV; pack_w_kernel<<<(n + 255) / 256, 256, 0, stream>>>(Wg,   wcat,  DD, DVV, COL_G);
  n = DVV * DD; pack_w_kernel<<<(n + 255) / 256, 256, 0, stream>>>(Wout, woutb, DVV, DD, 0);

  n = LL * DD;
  conv_silu_kernel<<<(n + 255) / 256, 256, 0, stream>>>(x, convw, xc);

  gemm_bf16_kernel<<<dim3(NCAT / 128, LL / 64), 256, 0, stream>>>(xc, wcat, qkvg, LL, NCAT, DD);

  router_kernel<<<LL * HH, 128, 0, stream>>>(qkvg, router_w, vpan, winr, wins);
  scan_kernel<<<HH, 1, 0, stream>>>(winr, wins, coef);
  build_qk_kernel<<<LL * HH, 64, 0, stream>>>(qkvg, winr, coef, qpan, kpan);

  attention_kernel<<<dim3(LL / 64, HH), 128, 0, stream>>>(qpan, kpan, vpan, obuf);

  ln_gate_kernel<<<LL * HH, 128, 0, stream>>>(obuf, qkvg, ybf);

  gemm_bf16_kernel<<<dim3(DD / 128, LL / 64), 256, 0, stream>>>(ybf, woutb, out, LL, DD, DVV);
}
